// LinearAttention3D_30537217474649
// MI455X (gfx1250) — matrix operand order repaired and checked by reading, not compiled, not run on hardware
//
#include <hip/hip_runtime.h>

// ---------------- problem constants (from reference) ----------------
#define B_    2
#define CIN   64      // input dim
#define DEP   32      // depth == e axis
#define NSP   512     // n = h*w
#define SSP   16384   // DEP*NSP spatial per batch
#define HEADS 4
#define DH    32      // dim_head
#define HID   128     // HEADS*DH
#define OC    64      // output channels
#define CNT_PER_B 16777216.0  // OC*NSP*NSP
#define QSCALE 0.17677669529663687f  // 32^-0.5

typedef __attribute__((ext_vector_type(16))) __bf16 v16bf;
typedef __attribute__((ext_vector_type(8)))  float  v8f;

// ---- WMMA fragment helpers (contiguous K layouts used by this code, wave32) ----
// A (16-bit, 16x32 MxK): lane L: M=L%16, hi=L/16; element j: K=16*hi+j
// (same K order as B).  'row' points at K-contiguous row data.
__device__ __forceinline__ v16bf afrag_bf16(const __bf16* row, int hi8) {
  v16bf a;
#pragma unroll
  for (int j = 0; j < 8; ++j) {
    int k = 2 * j + 2 * hi8;
    a[2 * j]     = row[k];
    a[2 * j + 1] = row[k + 1];
  }
  return a;
}
__device__ __forceinline__ v16bf afrag_f32(const float* row, int hi8) {
  v16bf a;
#pragma unroll
  for (int j = 0; j < 8; ++j) {
    int k = 2 * j + 2 * hi8;
    a[2 * j]     = (__bf16)row[k];
    a[2 * j + 1] = (__bf16)row[k + 1];
  }
  return a;
}
// B (16-bit, 32x16 KxN): lane L: N=L%16, element j: K=16*(L/16)+j.
// With K contiguous in memory this is one 32-byte load.

// 32-bit wave-relative LDS byte offset of a generic pointer into __shared__.
__device__ __forceinline__ unsigned lds_addr_of(const void* p) {
  return (unsigned)(unsigned long long)
      (__attribute__((address_space(3))) const char*)(const char*)p;
}

// ---------------- kernel 1: qkv projection ----------------
// qf[b][hc][s], kf[b][hc][s] fp32 ; vbf[b][hc][m][e] bf16 (e contiguous)
__global__ __launch_bounds__(256) void qkv_kernel(const float* __restrict__ x,
                                                  const float* __restrict__ wqkv,
                                                  float* __restrict__ qf,
                                                  float* __restrict__ kf,
                                                  __bf16* __restrict__ vbf) {
  const int s = blockIdx.x * 256 + threadIdx.x;
  const int b = blockIdx.y;
  float xr[CIN];
#pragma unroll
  for (int c = 0; c < CIN; ++c) xr[c] = x[(b * CIN + c) * SSP + s];
  const int e = s >> 9, m = s & 511;
  for (int o = 0; o < 3 * HID; ++o) {
    const float* wr = wqkv + o * CIN;
    float acc = 0.f;
#pragma unroll
    for (int c = 0; c < CIN; ++c) acc = fmaf(wr[c], xr[c], acc);
    if (o < HID) {
      qf[(b * HID + o) * SSP + s] = acc;
    } else if (o < 2 * HID) {
      kf[(b * HID + (o - HID)) * SSP + s] = acc;
    } else {
      int hc = o - 2 * HID;
      vbf[((b * HID + hc) * NSP + m) * DEP + e] = (__bf16)acc;
    }
  }
}

// ---------------- kernel 2: softmax over depth e for q, fold scale ----------------
// qbf[b][h][n][c][e] bf16 (e contiguous -> B-fragment friendly)
__global__ __launch_bounds__(256) void qsoftmax_kernel(const float* __restrict__ qf,
                                                       __bf16* __restrict__ qbf) {
  const int g = blockIdx.x * 256 + threadIdx.x;  // 2*128*512 threads
  const int b = g >> 16, hc = (g >> 9) & 127, n = g & 511;
  const float* base = qf + (b * HID + hc) * SSP + n;
  float v[DEP];
  float mx = -3.4e38f;
#pragma unroll
  for (int e = 0; e < DEP; ++e) { v[e] = base[e * NSP]; mx = fmaxf(mx, v[e]); }
  float sum = 0.f;
#pragma unroll
  for (int e = 0; e < DEP; ++e) { v[e] = expf(v[e] - mx); sum += v[e]; }
  const float sc = QSCALE / sum;
  const int h = hc >> 5, c = hc & 31;
  __bf16* out = qbf + (((b * HEADS + h) * NSP + n) * DH + c) * DEP;
#pragma unroll
  for (int e = 0; e < DEP; ++e) out[e] = (__bf16)(v[e] * sc);
}

// ---------------- kernel 3: softmax over n for k ----------------
// kbf[b][hc][n][e] bf16 (e contiguous -> A-fragment friendly)
__global__ __launch_bounds__(256) void ksoftmax_kernel(const float* __restrict__ kf,
                                                       __bf16* __restrict__ kbf) {
  const int g = blockIdx.x * 256 + threadIdx.x;  // 2*128*32 threads
  const int b = g >> 12, hc = (g >> 5) & 127, e = g & 31;
  const float* base = kf + (b * HID + hc) * SSP + e * NSP;
  float mx = -3.4e38f;
  for (int n = 0; n < NSP; ++n) mx = fmaxf(mx, base[n]);
  float sum = 0.f;
  for (int n = 0; n < NSP; ++n) sum += expf(base[n] - mx);
  const float inv = 1.f / sum;
  __bf16* out = kbf + (b * HID + hc) * NSP * DEP + e;
  for (int n = 0; n < NSP; ++n) out[n * DEP] = (__bf16)(expf(base[n] - mx) * inv);
}

// ---------------- kernel 4: fused wq / context / output GEMMs ----------------
// grid (8 m-tiles of 64, 32 n-tiles of 16, 2 batches), 512 threads = 16 waves.
// Wave w owns n_local = w.  LDS: wq[16n][64o][32c] (64KB) + ctx[16n][64m][32c] (64KB)
// + async-staged v tile vt[32c][64m][32e] (128KB) = 256KB dynamic LDS.
__global__ __launch_bounds__(512) void attn_kernel(const __bf16* __restrict__ qbf,
                                                   const __bf16* __restrict__ kbf,
                                                   const __bf16* __restrict__ vbf,
                                                   const float* __restrict__ wout,
                                                   const float* __restrict__ bout,
                                                   float* __restrict__ y,
                                                   double* __restrict__ stats) {
  extern __shared__ char smem[];
  __bf16* wq = (__bf16*)smem;                 // [16][64][32]
  __bf16* cx = (__bf16*)(smem + 65536);       // [16][64][32]
  __bf16* vt = (__bf16*)(smem + 131072);      // [32][64][32]

  const int t = threadIdx.x;
  const int wv = t >> 5, lane = t & 31, li = lane & 15, hi = lane >> 4;
  const int hi8 = hi * 8;
  const int b = blockIdx.z, nt = blockIdx.y, m0 = blockIdx.x * 64;
  const int n = nt * 16 + wv;
  const unsigned vt_base = lds_addr_of(vt);

  v8f acc[16] = {};

  for (int h = 0; h < HEADS; ++h) {
    // ---- stage 0: async-stage this wave's two v rows (c=2wv,2wv+1) into LDS.
    // 4KB contiguous per hc; 8 x b128 per lane-set; overlaps with stage 1 WMMAs.
#pragma unroll
    for (int cc = 0; cc < 2; ++cc) {
      const int c = wv * 2 + cc, hc = h * DH + c;
      const unsigned long long g =
          (unsigned long long)(vbf + ((size_t)(b * HID + hc) * NSP + m0) * DEP) +
          (unsigned)lane * 16u;
      const unsigned l = vt_base + (unsigned)c * 4096u + (unsigned)lane * 16u;
#pragma unroll
      for (int i = 0; i < 8; ++i) {
        asm volatile("global_load_async_to_lds_b128 %0, %1, off"
                     :: "v"(l + (unsigned)i * 512u),
                        "v"(g + (unsigned long long)i * 512ull)
                     : "memory");
      }
    }

    // ---- stage 1: wq[n][o][c] = sum_e w_out[o, h*32+e] * q[h,c,e,n]; wave wv does n=wv
#pragma unroll
    for (int ot = 0; ot < 4; ++ot) {
      const float* wr = wout + (ot * 16 + li) * HID + h * DH;
      v16bf a = afrag_f32(wr, hi8);
#pragma unroll
      for (int ct = 0; ct < 2; ++ct) {
        const __bf16* qp = qbf + ((b * HEADS + h) * NSP + n) * (DH * DEP) +
                           (ct * 16 + li) * DEP + hi * 16;
        v16bf bb = *(const v16bf*)qp;
        v8f d = {};
        d = __builtin_amdgcn_wmma_f32_16x16x32_bf16(false, a, false, bb,
                                                    (short)0, d, false, false);
#pragma unroll
        for (int v = 0; v < 8; ++v)
          wq[(wv * 64 + ot * 16 + v + hi8) * 32 + ct * 16 + li] = (__bf16)d[v];
      }
    }
    __syncthreads();

    // drain the async copies this wave issued (it consumes only its own chunks)
    asm volatile("s_wait_asynccnt 0x0" ::: "memory");

    // ---- stage 2: ctx[n][m][c] = sum_e k[hc,e,n] * v[hc,e,m]; wave wv does c = 2wv,2wv+1
#pragma unroll
    for (int cc = 0; cc < 2; ++cc) {
      const int c = wv * 2 + cc, hc = h * DH + c;
      const __bf16* kp = kbf + ((b * HID + hc) * NSP + nt * 16 + li) * DEP;
      v16bf a = afrag_bf16(kp, hi8);
#pragma unroll
      for (int mt = 0; mt < 4; ++mt) {
        const __bf16* vp = vt + c * 2048 + (mt * 16 + li) * DEP + hi * 16;
        v16bf bb = *(const v16bf*)vp;
        v8f d = {};
        d = __builtin_amdgcn_wmma_f32_16x16x32_bf16(false, a, false, bb,
                                                    (short)0, d, false, false);
#pragma unroll
        for (int v = 0; v < 8; ++v)
          cx[((v + hi8) * 64 + mt * 16 + li) * 32 + c] = (__bf16)d[v];
      }
    }
    __syncthreads();

    // ---- stage 3: y[o, n=wv, m] += wq[n][o][:] x ctx[n][:][m]   (K = c = 32)
#pragma unroll
    for (int ot = 0; ot < 4; ++ot) {
      const __bf16* ap = wq + (wv * 64 + ot * 16 + li) * 32;
      v16bf a = afrag_bf16(ap, hi8);
#pragma unroll
      for (int mt = 0; mt < 4; ++mt) {
        const __bf16* bp = cx + (wv * 64 + mt * 16 + li) * 32 + hi * 16;
        v16bf bb = *(const v16bf*)bp;
        acc[ot * 4 + mt] = __builtin_amdgcn_wmma_f32_16x16x32_bf16(
            false, a, false, bb, (short)0, acc[ot * 4 + mt], false, false);
      }
    }
    __syncthreads();
  }

  // ---- epilogue: write pre-norm y, accumulate batch sum / sumsq (bias folded)
  double sum = 0.0, sq = 0.0;
#pragma unroll
  for (int ot = 0; ot < 4; ++ot)
#pragma unroll
    for (int mt = 0; mt < 4; ++mt)
#pragma unroll
      for (int v = 0; v < 8; ++v) {
        const int o = ot * 16 + v + hi8;
        const int m = m0 + mt * 16 + li;
        const float val = acc[ot * 4 + mt][v];
        y[((b * OC + o) * NSP + n) * NSP + m] = val;
        const double sv = (double)val + (double)bout[o];
        sum += sv;
        sq += sv * sv;
      }
  __syncthreads();
  double* red = (double*)smem;  // 1024 doubles, fits in reused LDS
  red[t] = sum;
  red[t + 512] = sq;
  __syncthreads();
  for (int s = 256; s > 0; s >>= 1) {
    if (t < s) { red[t] += red[t + s]; red[t + 512] += red[t + 512 + s]; }
    __syncthreads();
  }
  if (t == 0) {
    atomicAdd(&stats[b * 2 + 0], red[0]);
    atomicAdd(&stats[b * 2 + 1], red[512]);
  }
}

// ---------------- kernel 5: in-place GroupNorm(1) + bias + affine ----------------
__global__ __launch_bounds__(256) void norm_kernel(const float* __restrict__ bout,
                                                   const float* __restrict__ gamma,
                                                   const float* __restrict__ beta,
                                                   const double* __restrict__ stats,
                                                   float* __restrict__ y) {
  const int g = blockIdx.x * 256 + threadIdx.x;
  const int b = g >> 24;
  const int o = (g >> 18) & 63;
  const double mu = stats[b * 2 + 0] * (1.0 / CNT_PER_B);
  const double var = stats[b * 2 + 1] * (1.0 / CNT_PER_B) - mu * mu;
  const float inv = rsqrtf((float)var + 1e-5f);
  const float val = y[g] + bout[o];
  y[g] = (val - (float)mu) * inv * gamma[o] + beta[o];
}

// ---------------- host side ----------------
extern "C" void kernel_launch(void* const* d_in, const int* in_sizes, int n_in,
                              void* d_out, int out_size, void* d_ws, size_t ws_size,
                              hipStream_t stream) {
  const float* x    = (const float*)d_in[0];
  const float* wqkv = (const float*)d_in[1];
  const float* wout = (const float*)d_in[2];
  const float* bout = (const float*)d_in[3];
  const float* gam  = (const float*)d_in[4];
  const float* bet  = (const float*)d_in[5];
  float* y = (float*)d_out;

  char* ws = (char*)d_ws;
  float*  qf    = (float*)(ws + 0);                 // 16,777,216 B
  float*  kf    = (float*)(ws + 16777216);          // 16,777,216 B
  __bf16* vbf   = (__bf16*)(ws + 33554432);         //  8,388,608 B
  __bf16* qbf   = (__bf16*)(ws + 41943040);         //  8,388,608 B
  __bf16* kbf   = (__bf16*)(ws + 50331648);         //  8,388,608 B
  double* stats = (double*)(ws + 58720256);         //         32 B

  hipMemsetAsync(stats, 0, 4 * sizeof(double), stream);

  qkv_kernel<<<dim3(SSP / 256, B_), 256, 0, stream>>>(x, wqkv, qf, kf, vbf);
  qsoftmax_kernel<<<(B_ * HID * NSP) / 256, 256, 0, stream>>>(qf, qbf);
  ksoftmax_kernel<<<(B_ * HID * DEP) / 256, 256, 0, stream>>>(kf, kbf);
  attn_kernel<<<dim3(NSP / 64, NSP / 16, B_), 512, 262144, stream>>>(
      qbf, kbf, vbf, wout, bout, y, stats);
  norm_kernel<<<(B_ * OC * NSP * NSP) / 256, 256, 0, stream>>>(bout, gam, bet, stats, y);
}
